// SeqAdder_63282048139425
// MI455X (gfx1250) — compile-verified, matrix operand order checked by reading, not run on hardware
//
#include <hip/hip_runtime.h>
#include <hip/hip_bf16.h>
#include <float.h>

// CDNA5 (gfx1250) wave32 WMMA implementation of the sequential digit-adder scan.
// Each wave owns 16 batch rows; the whole recurrence runs in-register with two
// v_wmma_f32_16x16x32_f16 per step and zero cross-lane data movement.

typedef __attribute__((ext_vector_type(16))) _Float16 v16h;
typedef __attribute__((ext_vector_type(8)))  _Float16 v8h;
typedef __attribute__((ext_vector_type(8)))  float    v8f;

__device__ __forceinline__ float sigmoidf_fast(float x) {
#if __has_builtin(__builtin_amdgcn_tanhf)
    // gfx1250 v_tanh_f32: sigmoid(x) = 0.5*tanh(x/2) + 0.5
    // (compiler fuses the fma + f16 convert into v_fma_mixlo_f16 when the
    //  result feeds a f16 matrix slot)
    return __builtin_fmaf(0.5f, __builtin_amdgcn_tanhf(0.5f * x), 0.5f);
#else
    // raw v_rcp_f32 instead of IEEE divide (no div_scale/div_fixup chain)
    return __builtin_amdgcn_rcpf(1.0f + __expf(-x));
#endif
}

__global__ __launch_bounds__(32)
void SeqAdder_wmma_kernel(const float* __restrict__ x1,
                          const float* __restrict__ x2,
                          const float* __restrict__ W1,
                          const float* __restrict__ b1,
                          const float* __restrict__ W2,
                          const float* __restrict__ b2,
                          float* __restrict__ out,
                          int Bc, int Lc) {
    const int  lane = threadIdx.x;     // 0..31 (wave32)
    const int  m    = lane & 15;       // matrix row owned by this lane
    const bool hi   = lane >= 16;      // hi half carries K=8..15 / K=24..31 slots
    const int  row  = blockIdx.x * 16 + m;

    // ---- Constant A-matrices (weights), built once per wave ------------------
    // A1 = W1^T padded to 16x32 f16; K=0..2 are W1 rows, K=3 carries b1
    // (bias folded via constant-1 input row). Lanes>=16 (K=8..15,24..31) = 0,
    // which makes hi-lane garbage in B1 harmless (multiplied by zero).
    v16h A1 = {};
    if (!hi) {
        A1[0] = (_Float16)W1[0 * 16 + m];
        A1[1] = (_Float16)W1[1 * 16 + m];
        A1[2] = (_Float16)W1[2 * 16 + m];
        A1[3] = (_Float16)b1[m];
    }
    // A2 = W2^T padded to 16x32 f16; K=0..15 = W2 rows (only output rows m<2
    // are nonzero), K=16 carries b2 (paired with constant-1 row in B2).
    v16h A2 = {};
    {
        const int klo = hi ? 8 : 0;
        if (m < 2) {
            #pragma unroll
            for (int j = 0; j < 8; ++j)
                A2[j] = (_Float16)W2[(klo + j) * 2 + m];
            if (!hi) A2[8] = (_Float16)b2[m];   // K=16 -> bias
        }
    }

    const float* x1row = x1 + (size_t)row * Lc;
    const float* x2row = x2 + (size_t)row * Lc;
    float*       yrow  = out + (size_t)row * Lc;

    const v8f zeroC = {};
    const v8h zeroH = {};
    float c = 0.0f;                     // carry recurrence (lanes 0..15 valid)

    // Software-pipelined streaming: loads for iteration l+4 issue before the
    // 4-step compute of iteration l, hiding VMEM latency behind the recurrence.
    float4 a4 = *(const float4*)(x1row);
    float4 b4 = *(const float4*)(x2row);

    for (int l = 0; l < Lc; l += 4) {
        float4 a4n, b4n;
        const bool more = (l + 4) < Lc;
        if (more) {
            a4n = *(const float4*)(x1row + l + 4);
            b4n = *(const float4*)(x2row + l + 4);
        }
        if (l + 256 < Lc) {             // stream ~1KB ahead into near caches
            __builtin_prefetch(x1row + l + 256, 0, 3);
            __builtin_prefetch(x2row + l + 256, 0, 3);
        }
        float4 y4;
        #pragma unroll
        for (int s = 0; s < 4; ++s) {
            const float a = (s == 0) ? a4.x : (s == 1) ? a4.y : (s == 2) ? a4.z : a4.w;
            const float b = (s == 0) ? b4.x : (s == 1) ? b4.y : (s == 2) ? b4.z : b4.w;

            // B1 = inp^T (32x16 f16): lane n holds K0=a, K1=b, K2=c, K3=1.
            v16h B1 = {};
            B1[0] = (_Float16)a;
            B1[1] = (_Float16)b;
            B1[2] = (_Float16)c;
            B1[3] = (_Float16)1.0f;

            // h^T (16 hid x 16 batch) in f32, bias already included via K=3.
            v8f D1 = __builtin_amdgcn_wmma_f32_16x16x32_f16(
                false, A1, false, B1, (short)0, zeroC, false, false);

            // f32->f16 pack then packed-f16 relu (relu commutes with the
            // monotone conversion): 4x v_cvt_pk_f16_f32 + 4x v_pk_max_num_f16.
            // D1's D-layout (VGPR r = hid r | hid r+8) is exactly the f16
            // B-matrix layout for layer 2 -> no lane shuffles.
            v8h h8;
            #pragma unroll
            for (int j = 0; j < 8; ++j)
                h8[j] = (_Float16)D1[j];
            h8 = __builtin_elementwise_max(h8, zeroH);

            v16h B2 = {};
            #pragma unroll
            for (int j = 0; j < 8; ++j)
                B2[j] = h8[j];
            B2[8] = (_Float16)1.0f;     // K=16 constant-1 row pairs with b2 in A2

            v8f D2 = __builtin_amdgcn_wmma_f32_16x16x32_f16(
                false, A2, false, B2, (short)0, zeroC, false, false);

            // out^T row 0 = sum bit, row 1 = carry (lanes 0..15 = batch n).
            const float y = sigmoidf_fast(D2[0]);
            c             = sigmoidf_fast(D2[1]);
            if      (s == 0) y4.x = y;
            else if (s == 1) y4.y = y;
            else if (s == 2) y4.z = y;
            else             y4.w = y;
        }
        if (!hi) *(float4*)(yrow + l) = y4;   // exec-masked store; WMMAs outside
        if (more) { a4 = a4n; b4 = b4n; }
    }
    if (!hi) out[(size_t)Bc * Lc + row] = c;  // c_final column
}

extern "C" void kernel_launch(void* const* d_in, const int* in_sizes, int n_in,
                              void* d_out, int out_size, void* d_ws, size_t ws_size,
                              hipStream_t stream) {
    const float* x1 = (const float*)d_in[0];
    const float* x2 = (const float*)d_in[1];
    const float* W1 = (const float*)d_in[2];   // [3,16] row-major
    const float* b1 = (const float*)d_in[3];   // [16]
    const float* W2 = (const float*)d_in[4];   // [16,2] row-major
    const float* b2 = (const float*)d_in[5];   // [2]
    float* out = (float*)d_out;

    const int L = 4096;
    const int B = in_sizes[0] / L;             // 4096

    dim3 grid(B / 16), block(32);              // 256 single-wave workgroups
    hipLaunchKernelGGL(SeqAdder_wmma_kernel, grid, block, 0, stream,
                       x1, x2, W1, b1, W2, b2, out, B, L);
}
